// Mamba2_49400713838841
// MI455X (gfx1250) — matrix operand order repaired and checked by reading, not compiled, not run on hardware
//
#include <hip/hip_runtime.h>
#include <hip/hip_bf16.h>

// ---------------------------------------------------------------------------
// MI455X (gfx1250) stereo Mamba-2 block.
// - All matrix math: v_wmma_f32_16x16x32_bf16 (bf16 in, f32 accumulate).
// - Projection GEMMs stage tiles with the Tensor Data Mover
//   (tensor_load_to_lds, TENSORcnt), double-buffered, with TDM LDS padding
//   reproducing the pitch-40 bf16 layout the WMMA fragment gathers expect.
// - Deterministic: fixed reduction orders, no float atomics.
// ---------------------------------------------------------------------------

typedef __bf16 bf16_t;
typedef __attribute__((ext_vector_type(8)))  __bf16 bf16x8;
typedef __attribute__((ext_vector_type(16))) __bf16 v16bf;
typedef __attribute__((ext_vector_type(8)))  float  v8f;
typedef unsigned int u32;
typedef __attribute__((ext_vector_type(4))) unsigned int u32x4;
typedef __attribute__((ext_vector_type(8))) int i32x8;
typedef __attribute__((ext_vector_type(4))) int i32x4;

#define D_MODEL   256
#define D_INNER   512
#define HEADDIM   64
#define NHEADS    8
#define D_STATE   64
#define D_IN_PROJ 1160     // 2*512 + 2*64 + 8
#define CONV_DIM  640      // 512 + 128
#define BATCH     4
#define SEQ       8192
#define HH        64
#define WWID      128
#define MTOT      (BATCH * SEQ)   // 32768
#define KV_ELEMS  (BATCH * NHEADS * 64 * 64)   // 131072
#define KV_CHUNKS 8
#define NPAD_PROJ 1216     // 19 * 64 (W_in / W_stereo rows padded)
#define NPAD_OUT  256

// Gather a 16-element bf16 fragment from two 16-byte-aligned LDS locations.
__device__ __forceinline__ v16bf ld_frag(const bf16_t* p0, const bf16_t* p1) {
  bf16x8 lo = *(const bf16x8*)p0;
  bf16x8 hi = *(const bf16x8*)p1;
  return __builtin_shufflevector(lo, hi, 0,1,2,3,4,5,6,7,8,9,10,11,12,13,14,15);
}

__device__ __forceinline__ v8f wmma_bf16(v16bf a, v16bf b, v8f c) {
  return __builtin_amdgcn_wmma_f32_16x16x32_bf16(false, a, false, b, (short)0, c,
                                                 false, false);
}

// ---------------------------------------------------------------------------
// TDM: async copy of a 2D bf16 tile (nrows x 32 elems) global->LDS.
// D# group0: count=1, lds_addr, 57-bit global addr, type=2.
// D# group1: data_size=2B, pad_enable, pad_interval=3 (every 64B),
//            pad_amount=3 (16B) -> LDS row pitch 80B == 40 bf16 elems.
// rowlen = tensor_dim0 = tensor_dim0_stride (elements); tile_dim0 = 32.
// ---------------------------------------------------------------------------
__device__ __forceinline__ void tdm_load_2d(u32 lds_off, const bf16_t* g,
                                            u32 rowlen, u32 nrows,
                                            u32 tensor_rows) {
  unsigned long long ga = (unsigned long long)(uintptr_t)g;
  u32x4 g0;
  g0.x = 1u;                                        // count=1 (user descriptor)
  g0.y = lds_off;                                   // lds_addr (bytes)
  g0.z = (u32)ga;                                   // global_addr[31:0]
  g0.w = ((u32)(ga >> 32) & 0x1FFFFFFu) | 0x80000000u;  // ga[56:32] | type=2
  i32x8 g1;
  g1[0] = (1 << 16) | (1 << 20) | (3 << 22) | (3 << 25);
  g1[1] = (int)((rowlen & 0xFFFFu) << 16);          // tensor_dim0 lo
  g1[2] = (int)((rowlen >> 16) | ((tensor_rows & 0xFFFFu) << 16));
  g1[3] = (int)((tensor_rows >> 16) | (32u << 16)); // tensor_dim1 hi | tile_dim0
  g1[4] = (int)nrows;                               // tile_dim1 (tile_dim2 = 0)
  g1[5] = (int)rowlen;                              // tensor_dim0_stride lo
  g1[6] = 0;
  g1[7] = 0;
  i32x4 z4 = {0, 0, 0, 0};
#if defined(__clang_major__) && (__clang_major__ >= 23)
  i32x8 z8 = {0, 0, 0, 0, 0, 0, 0, 0};
  __builtin_amdgcn_tensor_load_to_lds(g0, g1, z4, z4, z8, 0);
#else
  __builtin_amdgcn_tensor_load_to_lds(g0, g1, z4, z4, 0);
#endif
}

__device__ __forceinline__ u32 lds_addr32(const void* p) {
  return (u32)(uintptr_t)p;   // flat LDS aperture: low 32 bits = LDS offset
}

// ---------------------------------------------------------------------------
// GEMM: C[M,N] = A[M,K] (bf16 row-major) @ W[Npad,K]^T (bf16, rows >= N
// zero-padded). WG = 256 threads (8 waves); WG tile 128x64; wave tile 32x32.
// K-chunk 32, double-buffered TDM staging issued by wave 0.
// ---------------------------------------------------------------------------
__global__ __launch_bounds__(256) void gemm_tdm_kernel(
    const bf16_t* __restrict__ A, const bf16_t* __restrict__ W,
    float* __restrict__ C, int M, int N, int K) {
  __shared__ __align__(16) bf16_t sA[2][128 * 40];
  __shared__ __align__(16) bf16_t sB[2][64 * 40];
  const int t    = threadIdx.x;
  const int lane = t & 31, wave = t >> 5;
  const int lm   = lane & 15, lh = lane >> 4;
  const int wm   = wave & 3, wn = wave >> 2;
  const long blockM = (long)blockIdx.y * 128;
  const int  blockN = blockIdx.x * 64;
  const int  ktiles = K >> 5;

  const bf16_t* aBase = A + blockM * (long)K;
  const bf16_t* bBase = W + (long)blockN * K;

  if (t < 32) {  // wave 0 drives the Tensor Data Mover
    tdm_load_2d(lds_addr32(&sA[0][0]), aBase, (u32)K, 128u, (u32)M);
    tdm_load_2d(lds_addr32(&sB[0][0]), bBase, (u32)K, 64u, 0x100000u);
  }

  v8f acc[2][2] = {};

  for (int kt = 0; kt < ktiles; kt++) {
    if (t < 32) {
      if (kt + 1 < ktiles) {
        const int nb = (kt + 1) & 1;
        const int k0 = (kt + 1) * 32;
        tdm_load_2d(lds_addr32(&sA[nb][0]), aBase + k0, (u32)K, 128u, (u32)M);
        tdm_load_2d(lds_addr32(&sB[nb][0]), bBase + k0, (u32)K, 64u, 0x100000u);
        __builtin_amdgcn_s_wait_tensorcnt(2);  // current tile's 2 loads done
      } else {
        __builtin_amdgcn_s_wait_tensorcnt(0);
      }
    }
    __syncthreads();   // publish TDM-written LDS to all 8 waves

    const bf16_t* cA = sA[kt & 1];
    const bf16_t* cB = sB[kt & 1];
    v16bf af[2], bfv[2];
    #pragma unroll
    for (int ms = 0; ms < 2; ms++) {
      const bf16_t* pa = cA + (wm * 32 + ms * 16 + lm) * 40 + lh * 8;
      af[ms] = ld_frag(pa, pa + 16);
    }
    #pragma unroll
    for (int ns = 0; ns < 2; ns++) {
      const bf16_t* pb = cB + (wn * 32 + ns * 16 + lm) * 40 + lh * 8;
      bfv[ns] = ld_frag(pb, pb + 16);
    }
    #pragma unroll
    for (int ms = 0; ms < 2; ms++)
      #pragma unroll
      for (int ns = 0; ns < 2; ns++)
        acc[ms][ns] = wmma_bf16(af[ms], bfv[ns], acc[ms][ns]);
    __syncthreads();   // compute done before buffer is re-filled next iter
  }

  #pragma unroll
  for (int ms = 0; ms < 2; ms++)
    #pragma unroll
    for (int ns = 0; ns < 2; ns++) {
      const long row0 = blockM + wm * 32 + ms * 16 + lh * 8;
      const int  col  = blockN + wn * 32 + ns * 16 + lm;
      if (col < N) {
        #pragma unroll
        for (int i = 0; i < 8; i++)
          C[(row0 + i) * (long)N + col] = acc[ms][ns][i];
      }
    }
}

// ---------------------------------------------------------------------------
// One-time precision conversions.
// ---------------------------------------------------------------------------
__global__ __launch_bounds__(256) void cvt_a_kernel(
    const float* __restrict__ src, bf16_t* __restrict__ dst, int n) {
  const int i = blockIdx.x * 256 + threadIdx.x;
  if (i < n) dst[i] = (bf16_t)src[i];
}

// Weights (N,K) -> bf16 (Npad,K), rows >= N zero-filled.
__global__ __launch_bounds__(256) void cvt_w_kernel(
    const float* __restrict__ W, bf16_t* __restrict__ out,
    int N, int K, int Npad) {
  const int i = blockIdx.x * 256 + threadIdx.x;
  if (i >= Npad * K) return;
  const int n = i / K;
  out[i] = (n < N) ? (bf16_t)W[i] : (bf16_t)0.f;
}

// ---------------------------------------------------------------------------
// dt = softplus(zx[:, 1152+h] + dt_bias[h])
// ---------------------------------------------------------------------------
__global__ __launch_bounds__(256) void dt_kernel(
    const float* __restrict__ zx, const float* __restrict__ dt_bias,
    float* __restrict__ dtb) {
  const int i = blockIdx.x * 256 + threadIdx.x;
  if (i >= MTOT * NHEADS) return;
  const int h = i & 7;
  const long row = i >> 3;
  const float x = zx[row * D_IN_PROJ + (D_INNER + CONV_DIM) + h] + dt_bias[h];
  dtb[i] = (x > 20.f) ? x : log1pf(__expf(x));
}

// ---------------------------------------------------------------------------
// Depthwise 3x3 SAME conv + bias + SiLU over the xBC slice of zx.
// ---------------------------------------------------------------------------
__global__ __launch_bounds__(128) void conv_silu_kernel(
    const float* __restrict__ zx,
    const float* __restrict__ wl, const float* __restrict__ bl,
    const float* __restrict__ wr, const float* __restrict__ br,
    float* __restrict__ out) {
  const int c = blockIdx.x * 128 + threadIdx.x;      // < 640
  const int p = blockIdx.y;
  const int b = blockIdx.z;
  const int h = p >> 7, w = p & 127;
  const float* wt = (b < 2) ? wl : wr;
  const float* bs = (b < 2) ? bl : br;
  const float* in = zx + (long)b * SEQ * D_IN_PROJ + D_INNER;

  float acc = bs[c];
  #pragma unroll
  for (int kh = 0; kh < 3; kh++) {
    const int hh = h + kh - 1;
    if ((unsigned)hh >= (unsigned)HH) continue;
    #pragma unroll
    for (int kw = 0; kw < 3; kw++) {
      const int ww = w + kw - 1;
      if ((unsigned)ww >= (unsigned)WWID) continue;
      acc += wt[c * 9 + kh * 3 + kw] *
             in[((long)hh * WWID + ww) * D_IN_PROJ + c];
    }
  }
  const float sv = acc / (1.f + __expf(-acc));   // SiLU
  out[((long)b * SEQ + p) * CONV_DIM + c] = sv;
}

// ---------------------------------------------------------------------------
// Pass 1: partial KV[s,p] = sum_l B[l,s] * (x[l,h,p] * dt[l,h]*exp(A_log[h]))
// ---------------------------------------------------------------------------
__global__ __launch_bounds__(256) void attn_kv_kernel(
    const float* __restrict__ xc, const float* __restrict__ dtb,
    const float* __restrict__ A_log, float* __restrict__ kvpart) {
  __shared__ __align__(16) bf16_t sB[64 * 40];   // [s][l]
  __shared__ __align__(16) bf16_t sX[64 * 40];   // [p][l]
  const int t    = threadIdx.x;
  const int lane = t & 31, wave = t >> 5;
  const int lm   = lane & 15, lh = lane >> 4;
  const int srow = wave & 3, ph = wave >> 2;
  const int h = blockIdx.y, b = blockIdx.z;
  const long base  = (long)b * SEQ;
  const int  lbase = blockIdx.x * 1024;
  const float aexp = __expf(A_log[h]);

  v8f acc0 = {}, acc1 = {};
  const int cidx = t & 63;
  const int lq   = (t >> 6) * 8;

  for (int kk = 0; kk < 1024; kk += 32) {
    #pragma unroll
    for (int j = 0; j < 8; j++) {
      const int lk = lq + j;
      const long l = base + lbase + kk + lk;
      const float da = dtb[l * NHEADS + h] * aexp;
      const float xv = xc[l * CONV_DIM + h * HEADDIM + cidx];
      const float bv = xc[l * CONV_DIM + D_INNER + cidx];
      sX[cidx * 40 + lk] = (bf16_t)(xv * da);
      sB[cidx * 40 + lk] = (bf16_t)bv;
    }
    __syncthreads();
    const bf16_t* pa = sB + (srow * 16 + lm) * 40 + lh * 8;
    const v16bf a = ld_frag(pa, pa + 16);
    const bf16_t* pb0 = sX + (ph * 32 + lm) * 40 + lh * 8;
    const bf16_t* pb1 = sX + (ph * 32 + 16 + lm) * 40 + lh * 8;
    acc0 = wmma_bf16(a, ld_frag(pb0, pb0 + 16), acc0);
    acc1 = wmma_bf16(a, ld_frag(pb1, pb1 + 16), acc1);
    __syncthreads();
  }

  float* dst = kvpart + ((long)(blockIdx.x * BATCH + b) * NHEADS + h) * 4096;
  #pragma unroll
  for (int i = 0; i < 8; i++) {
    const int sg = srow * 16 + lh * 8 + i;
    dst[sg * 64 + ph * 32 + lm]      = acc0[i];
    dst[sg * 64 + ph * 32 + 16 + lm] = acc1[i];
  }
}

__global__ __launch_bounds__(256) void kv_reduce_kernel(
    const float* __restrict__ part, float* __restrict__ kv) {
  const int i = blockIdx.x * 256 + threadIdx.x;   // < KV_ELEMS
  float s = 0.f;
  #pragma unroll
  for (int c = 0; c < KV_CHUNKS; c++) s += part[(long)c * KV_ELEMS + i];
  kv[i] = s;
}

// ---------------------------------------------------------------------------
// Pass 2: y[l,h,p] = sum_s C[l,s]*KV[s,p] + x[l,h,p]*D[h]
// ---------------------------------------------------------------------------
__global__ __launch_bounds__(256) void attn_y_kernel(
    const float* __restrict__ xc, const float* __restrict__ kv,
    const float* __restrict__ Dl, const float* __restrict__ Dr,
    float* __restrict__ yraw) {
  __shared__ __align__(16) bf16_t sC[128 * 72];  // [l][s]
  __shared__ __align__(16) bf16_t sK[64 * 72];   // [p][s]
  const int t    = threadIdx.x;
  const int lane = t & 31, wave = t >> 5;
  const int lm   = lane & 15, lh = lane >> 4;
  const int h = blockIdx.y, b = blockIdx.z;
  const int lbase = blockIdx.x * 128;
  const long base = (long)b * SEQ;

  {
    const float* kvp = kv + ((long)(b * NHEADS + h)) * 4096;
    const int p = t & 63, s0 = (t >> 6) * 16;
    #pragma unroll
    for (int i = 0; i < 16; i++)
      sK[p * 72 + s0 + i] = (bf16_t)kvp[(s0 + i) * 64 + p];
  }
  {
    const int ll = t >> 1, sh = (t & 1) * 32;
    const float* src = xc + (base + lbase + ll) * CONV_DIM + (D_INNER + D_STATE) + sh;
    bf16_t* dst = sC + ll * 72 + sh;
    #pragma unroll
    for (int i = 0; i < 32; i += 4) {
      float4 v = *(const float4*)(src + i);
      dst[i+0] = (bf16_t)v.x; dst[i+1] = (bf16_t)v.y;
      dst[i+2] = (bf16_t)v.z; dst[i+3] = (bf16_t)v.w;
    }
  }
  __syncthreads();

  v8f acc[4] = {};
  #pragma unroll
  for (int s0 = 0; s0 < 64; s0 += 32) {
    const bf16_t* pa = sC + (wave * 16 + lm) * 72 + s0 + lh * 8;
    const v16bf a = ld_frag(pa, pa + 16);
    #pragma unroll
    for (int pt = 0; pt < 4; pt++) {
      const bf16_t* pb = sK + (pt * 16 + lm) * 72 + s0 + lh * 8;
      acc[pt] = wmma_bf16(a, ld_frag(pb, pb + 16), acc[pt]);
    }
  }

  const float Dh = (b < 2 ? Dl : Dr)[h];
  #pragma unroll
  for (int pt = 0; pt < 4; pt++) {
    #pragma unroll
    for (int i = 0; i < 8; i++) {
      const long l = base + lbase + wave * 16 + lh * 8 + i;
      const int  p = pt * 16 + lm;
      const float xv = xc[l * CONV_DIM + h * HEADDIM + p];
      yraw[l * D_INNER + h * HEADDIM + p] = acc[pt][i] + xv * Dh;
    }
  }
}

// ---------------------------------------------------------------------------
// LayerNorm over 512 channels, multiply by z, emit bf16 (next GEMM's A).
// ---------------------------------------------------------------------------
__global__ __launch_bounds__(256) void ln_mul_kernel(
    const float* __restrict__ yraw, const float* __restrict__ g,
    const float* __restrict__ bln, const float* __restrict__ zsrc,
    bf16_t* __restrict__ out) {
  const long row = blockIdx.x;
  const int  t   = threadIdx.x;
  const float* yr = yraw + row * D_INNER;
  const float v0 = yr[t], v1 = yr[t + 256];
  __shared__ float s1[256], s2[256];
  s1[t] = v0 + v1;
  s2[t] = v0 * v0 + v1 * v1;
  __syncthreads();
  for (int off = 128; off > 0; off >>= 1) {
    if (t < off) { s1[t] += s1[t + off]; s2[t] += s2[t + off]; }
    __syncthreads();
  }
  const float mean = s1[0] * (1.f / 512.f);
  const float var  = s2[0] * (1.f / 512.f) - mean * mean;
  const float rstd = rsqrtf(var + 1e-5f);
  const float* zp = zsrc + row * D_IN_PROJ;
  out[row * D_INNER + t] =
      (bf16_t)(((v0 - mean) * rstd * g[t] + bln[t]) * zp[t]);
  out[row * D_INNER + t + 256] =
      (bf16_t)(((v1 - mean) * rstd * g[t + 256] + bln[t + 256]) * zp[t + 256]);
}

// ---------------------------------------------------------------------------
extern "C" void kernel_launch(void* const* d_in, const int* in_sizes, int n_in,
                              void* d_out, int out_size, void* d_ws, size_t ws_size,
                              hipStream_t stream) {
  (void)in_sizes; (void)n_in; (void)out_size; (void)ws_size;
  const float* u       = (const float*)d_in[0];
  const float* W_in    = (const float*)d_in[3];
  const float* W_st    = (const float*)d_in[4];
  const float* conv_lw = (const float*)d_in[5];
  const float* conv_lb = (const float*)d_in[6];
  const float* conv_rw = (const float*)d_in[7];
  const float* conv_rb = (const float*)d_in[8];
  const float* conv_w  = (const float*)d_in[9];
  const float* conv_b  = (const float*)d_in[10];
  const float* dt_bias = (const float*)d_in[11];
  const float* A_log   = (const float*)d_in[12];
  const float* D_l     = (const float*)d_in[13];
  const float* D_r     = (const float*)d_in[14];
  const float* Dm      = (const float*)d_in[15];
  const float* ln_g    = (const float*)d_in[16];
  const float* ln_b    = (const float*)d_in[17];
  const float* W_out   = (const float*)d_in[18];
  float* out = (float*)d_out;

  float* ws   = (float*)d_ws;
  float* zx   = ws;                                    // 32768*1160 f32
  float* xc   = zx   + (long)MTOT * D_IN_PROJ;         // 32768*640 f32
  float* dtb  = xc   + (long)MTOT * CONV_DIM;          // 32768*8 f32
  float* kvp  = dtb  + (long)MTOT * NHEADS;            // 8*131072 f32
  float* kvb  = kvp  + (long)KV_CHUNKS * KV_ELEMS;     // 131072 f32
  float* yraw = kvb  + (long)KV_ELEMS;                 // 32768*512 f32
  bf16_t* ubf = (bf16_t*)(yraw + (long)MTOT * D_INNER);// 32768*256 bf16
  bf16_t* ybf = ubf + (long)MTOT * D_MODEL;            // 32768*512 bf16
  bf16_t* wbf_in  = ybf + (long)MTOT * D_INNER;        // 1216*256 bf16
  bf16_t* wbf_st  = wbf_in + (long)NPAD_PROJ * D_MODEL;// 1216*512 bf16
  bf16_t* wbf_out = wbf_st + (long)NPAD_PROJ * D_INNER;// 256*512 bf16

  const dim3 blk256(256);
  const dim3 gProj(NPAD_PROJ / 64, MTOT / 128);
  const dim3 gOut(NPAD_OUT / 64, MTOT / 128);
  const dim3 gConv(5, SEQ, BATCH);
  const dim3 gKV(KV_CHUNKS, NHEADS, BATCH);
  const dim3 gY(SEQ / 128, NHEADS, BATCH);

  // ---- One-time bf16 conversions ----
  cvt_a_kernel<<<(MTOT * D_MODEL) / 256, blk256, 0, stream>>>(u, ubf, MTOT * D_MODEL);
  cvt_w_kernel<<<(NPAD_PROJ * D_MODEL) / 256, blk256, 0, stream>>>(
      W_in, wbf_in, D_IN_PROJ, D_MODEL, NPAD_PROJ);
  cvt_w_kernel<<<(NPAD_PROJ * D_INNER) / 256, blk256, 0, stream>>>(
      W_st, wbf_st, D_IN_PROJ, D_INNER, NPAD_PROJ);
  cvt_w_kernel<<<(NPAD_OUT * D_INNER) / 256, blk256, 0, stream>>>(
      W_out, wbf_out, D_MODEL, D_INNER, NPAD_OUT);

  // ---- Stage 1 ----
  gemm_tdm_kernel<<<gProj, blk256, 0, stream>>>(ubf, wbf_in, zx, MTOT, D_IN_PROJ, D_MODEL);
  dt_kernel<<<(MTOT * NHEADS) / 256, blk256, 0, stream>>>(zx, dt_bias, dtb);
  conv_silu_kernel<<<gConv, 128, 0, stream>>>(zx, conv_lw, conv_lb, conv_rw, conv_rb, xc);
  attn_kv_kernel<<<gKV, blk256, 0, stream>>>(xc, dtb, A_log, kvp);
  kv_reduce_kernel<<<KV_ELEMS / 256, blk256, 0, stream>>>(kvp, kvb);
  attn_y_kernel<<<gY, blk256, 0, stream>>>(xc, kvb, D_l, D_r, yraw);
  ln_mul_kernel<<<MTOT, blk256, 0, stream>>>(yraw, ln_g, ln_b, zx, ybf);

  // ---- Stage 2 (stereo mix) ----
  gemm_tdm_kernel<<<gProj, blk256, 0, stream>>>(ybf, wbf_st, zx, MTOT, D_IN_PROJ, D_INNER);
  dt_kernel<<<(MTOT * NHEADS) / 256, blk256, 0, stream>>>(zx, dt_bias, dtb);
  conv_silu_kernel<<<gConv, 128, 0, stream>>>(zx, conv_w, conv_b, conv_w, conv_b, xc);
  attn_kv_kernel<<<gKV, blk256, 0, stream>>>(xc, dtb, A_log, kvp);
  kv_reduce_kernel<<<KV_ELEMS / 256, blk256, 0, stream>>>(kvp, kvb);
  attn_y_kernel<<<gY, blk256, 0, stream>>>(xc, kvb, Dm, Dm, yraw);
  ln_mul_kernel<<<MTOT, blk256, 0, stream>>>(yraw, ln_g, ln_b, zx, ybf);

  // ---- Output projection ----
  gemm_tdm_kernel<<<gOut, blk256, 0, stream>>>(ybf, wbf_out, out, MTOT, D_MODEL, D_INNER);
}
